// TransformerFFNAttention1D_7988639170891
// MI455X (gfx1250) — matrix operand order repaired and checked by reading, not compiled, not run on hardware
//
#include <hip/hip_runtime.h>
#include <hip/hip_bf16.h>
#include <math.h>

// ---------------------------------------------------------------------------
// TransformerFFNAttention1D block for MI455X (gfx1250, wave32, WMMA).
// All four big GEMMs (mix 512x512, out 512x512, ffn1 512->2048, ffn2 2048->512,
// M = B*L = 32768; ~172 GFLOP total) run through v_wmma_f32_16x16x32_bf16 with
// f32 accumulation. Tiles are staged global->LDS with the CDNA5 async-LDS DMA
// path (GLOBAL_LOAD_ASYNC_TO_LDS_B128, ASYNCcnt) in a double-buffered pipeline
// so WMMA overlaps the next stage's fetch. ~0.5 GB of intermediate traffic at
// 23.3 TB/s is balanced against the WMMA time, so both paths matter.
// ---------------------------------------------------------------------------

typedef __bf16 bf16_t;
typedef __attribute__((ext_vector_type(16))) __bf16 v16bf;
typedef __attribute__((ext_vector_type(8)))  __bf16 v8bf;
typedef __attribute__((ext_vector_type(8)))  float  v8f;

#define DIMC     512
#define SEQ      2048
#define BATCH    16
#define HIDF     2048
#define GHID     128
#define NROWS    (BATCH * SEQ)          // 32768
#define EPSLN    1e-5f

__device__ __forceinline__ float gelu_exact(float v) {
    return 0.5f * v * (1.0f + erff(v * 0.70710678118654752f));
}

// Low 32 bits of a generic pointer to LDS == LDS byte address (aperture rule).
__device__ __forceinline__ unsigned lds_addr(const void* p) {
    return (unsigned)(unsigned long long)(p);
}

// CDNA5 async DMA: 16 bytes global -> LDS, tracked by ASYNCcnt (no VGPR data).
__device__ __forceinline__ void async_load_b128(unsigned lds, const void* gaddr) {
    asm volatile("global_load_async_to_lds_b128 %0, %1, off"
                 :: "v"(lds), "v"(gaddr) : "memory");
}

// ---------------------------------------------------------------------------
// LayerNorm over D=512 per row, output bf16. One 256-thread block per row.
// ---------------------------------------------------------------------------
__global__ void ln_rows_kernel(const float* __restrict__ x,
                               const float* __restrict__ g,
                               const float* __restrict__ b,
                               bf16_t* __restrict__ out) {
    __shared__ float red[256];
    size_t row = blockIdx.x;
    const float* xr = x + row * DIMC;
    int t = threadIdx.x;
    float v0 = xr[t], v1 = xr[t + 256];
    red[t] = v0 + v1;
    __syncthreads();
    for (int o = 128; o > 0; o >>= 1) { if (t < o) red[t] += red[t + o]; __syncthreads(); }
    float mean = red[0] * (1.0f / DIMC);
    __syncthreads();
    float d0 = v0 - mean, d1 = v1 - mean;
    red[t] = d0 * d0 + d1 * d1;
    __syncthreads();
    for (int o = 128; o > 0; o >>= 1) { if (t < o) red[t] += red[t + o]; __syncthreads(); }
    float rstd = rsqrtf(red[0] * (1.0f / DIMC) + EPSLN);
    bf16_t* orow = out + row * DIMC;
    orow[t]       = (bf16_t)(d0 * rstd * g[t]       + b[t]);
    orow[t + 256] = (bf16_t)(d1 * rstd * g[t + 256] + b[t + 256]);
}

// ---------------------------------------------------------------------------
// Fused (dwconv3 + dwconv5 + dwconv7)/3 + bias + exact GELU (conv along L,
// [B,L,D] layout so D is coalesced; cross-correlation like lax.conv).
// ---------------------------------------------------------------------------
__global__ void conv_gelu_kernel(const bf16_t* __restrict__ h,
                                 const float* __restrict__ w3, const float* __restrict__ b3,
                                 const float* __restrict__ w5, const float* __restrict__ b5,
                                 const float* __restrict__ w7, const float* __restrict__ b7,
                                 bf16_t* __restrict__ gout) {
    size_t idx = (size_t)blockIdx.x * blockDim.x + threadIdx.x;   // over B*L*D
    if (idx >= (size_t)NROWS * DIMC) return;
    int d = (int)(idx % DIMC);
    size_t bl = idx / DIMC;
    int l = (int)(bl % SEQ);
    float acc = 0.0f;
#pragma unroll
    for (int o = -3; o <= 3; ++o) {
        float wt = w7[d * 7 + (o + 3)];
        if (o >= -2 && o <= 2) wt += w5[d * 5 + (o + 2)];
        if (o >= -1 && o <= 1) wt += w3[d * 3 + (o + 1)];
        int ll = l + o;
        float hv = (ll >= 0 && ll < SEQ) ? (float)h[idx + (ptrdiff_t)o * DIMC] : 0.0f;
        acc += wt * hv;
    }
    acc = (acc + b3[d] + b5[d] + b7[d]) * (1.0f / 3.0f);
    gout[idx] = (bf16_t)gelu_exact(acc);
}

// ---------------------------------------------------------------------------
// Weight prep: cast, and transpose [K][N] fp32 -> [N][K] bf16 so every GEMM's
// B fragment is a contiguous K-run per lane.
// ---------------------------------------------------------------------------
__global__ void cast_copy_kernel(const float* __restrict__ src, bf16_t* __restrict__ dst, int n) {
    int i = blockIdx.x * blockDim.x + threadIdx.x;
    if (i < n) dst[i] = (bf16_t)src[i];
}

__global__ void transpose_cast_kernel(const float* __restrict__ src, bf16_t* __restrict__ dst,
                                      int K, int N) {
    __shared__ float tile[32][33];
    int k0 = blockIdx.y * 32, n0 = blockIdx.x * 32;
    int tx = threadIdx.x, ty = threadIdx.y;   // block (32, 8)
#pragma unroll
    for (int i = 0; i < 32; i += 8) {
        int k = k0 + ty + i, n = n0 + tx;
        tile[ty + i][tx] = (k < K && n < N) ? src[(size_t)k * N + n] : 0.0f;
    }
    __syncthreads();
#pragma unroll
    for (int i = 0; i < 32; i += 8) {
        int n = n0 + ty + i, k = k0 + tx;
        if (n < N && k < K) dst[(size_t)n * K + k] = (bf16_t)tile[tx][ty + i];
    }
}

// ---------------------------------------------------------------------------
// channel_token = mean over L of mixed [B,L,D] -> ct [B,D]
// ---------------------------------------------------------------------------
__global__ void ct_mean_kernel(const float* __restrict__ mixed, float* __restrict__ ct) {
    int b = blockIdx.x;
    int d = blockIdx.y * 256 + threadIdx.x;
    const float* p = mixed + (size_t)b * SEQ * DIMC + d;
    float s = 0.0f;
    for (int l = 0; l < SEQ; ++l) s += p[(size_t)l * DIMC];
    ct[b * DIMC + d] = s * (1.0f / SEQ);
}

// ---------------------------------------------------------------------------
// gate = sigmoid(gelu(ct @ cg_w1 + cg_b1) @ cg_w2 + cg_b2)   (tiny: 16x512)
// ---------------------------------------------------------------------------
__global__ void gate_kernel(const float* __restrict__ ct,
                            const float* __restrict__ w1, const float* __restrict__ b1,
                            const float* __restrict__ w2, const float* __restrict__ b2,
                            float* __restrict__ gate) {
    __shared__ float t1[GHID];
    __shared__ float ctb[DIMC];
    int b = blockIdx.x, t = threadIdx.x;   // 128 threads
    for (int i = t; i < DIMC; i += GHID) ctb[i] = ct[b * DIMC + i];
    __syncthreads();
    float s = b1[t];
    for (int d = 0; d < DIMC; ++d) s += ctb[d] * w1[d * GHID + t];
    t1[t] = gelu_exact(s);
    __syncthreads();
    for (int d = t; d < DIMC; d += GHID) {
        float s2 = b2[d];
        for (int j = 0; j < GHID; ++j) s2 += t1[j] * w2[j * DIMC + d];
        gate[b * DIMC + d] = 1.0f / (1.0f + expf(-s2));
    }
}

// ---------------------------------------------------------------------------
// Ag[b,l,c] = bf16(mixed[b,l,c] * gate[b,c])  (A operand for the out-proj GEMM)
// ---------------------------------------------------------------------------
__global__ void gate_apply_kernel(const float* __restrict__ mixed,
                                  const float* __restrict__ gate,
                                  bf16_t* __restrict__ out) {
    size_t idx = (size_t)blockIdx.x * blockDim.x + threadIdx.x;   // B*L*D
    if (idx >= (size_t)NROWS * DIMC) return;
    int d = (int)(idx % DIMC);
    int b = (int)(idx / ((size_t)SEQ * DIMC));
    out[idx] = (bf16_t)(mixed[idx] * gate[b * DIMC + d]);
}

// ---------------------------------------------------------------------------
// WMMA GEMM: C[M,N] = A[M,K] * Bt[N,K]^T (+ epilogue), A/Bt bf16.
// Macro tile 64x128, BK=64 (two k=32 substeps), 256 threads = 8 waves
// (4 in M x 2 in N); each wave owns a 16x64 strip -> 8 wmma per stage.
// Double-buffered LDS, filled with GLOBAL_LOAD_ASYNC_TO_LDS_B128 (6 per thread
// per stage); s_wait_asynccnt 6 after issuing stage k+1 guarantees stage k
// landed (async loads complete in order), then one workgroup barrier.
// Fragment layouts per CDNA5 ISA:
//   A lane l (<16): M=l,  K in {kb..kb+7, kb+16..kb+23}, kb=0 ; l>=16: kb=8
//   B lane l (<16): N=l,  K in {kb..kb+7, kb+16..kb+23}, kb=0 ; l>=16: N=l-16, kb=8
//   C lane l: N=l&15, VGPR r -> M = r + (l<16 ? 0 : 8)
// MODE 1: out f32 = acc + bias
// MODE 2: out f32 = resid + (acc + bias)*scale
// MODE 3: out bf16 = gelu(acc + bias)
// ---------------------------------------------------------------------------
template <int MODE, int N_DIM, int K_DIM>
__global__ __launch_bounds__(256)
void gemm_wmma_kernel(const bf16_t* __restrict__ A,
                      const bf16_t* __restrict__ Bt,
                      const float* __restrict__ bias,
                      const float* __restrict__ resid,
                      const float* __restrict__ scale,
                      void* __restrict__ Out) {
    constexpr int BM = 64, BN = 128, BK = 64;
    constexpr int NK = K_DIM / BK;
    __shared__ alignas(16) bf16_t As[2][BM][BK];
    __shared__ alignas(16) bf16_t Bs[2][BN][BK];

    const int m0  = blockIdx.y * BM;
    const int n0  = blockIdx.x * BN;
    const int tid = threadIdx.x;
    const int wave = tid >> 5, lane = tid & 31;
    const int wm = wave & 3, wn = wave >> 2;

    // Per-thread staging coordinates (whole tile covered by 256 threads).
    const int arow = tid >> 2, acol = (tid & 3) * 16;   // A: 2 x b128
    const int brow = tid >> 1, bcol = (tid & 1) * 32;   // B: 4 x b128

    auto issue_stage = [&](int kblk, int buf) {
        const int k0 = kblk * BK;
        const bf16_t* ga = A  + (size_t)(m0 + arow) * K_DIM + k0 + acol;
        async_load_b128(lds_addr(&As[buf][arow][acol]),      ga);
        async_load_b128(lds_addr(&As[buf][arow][acol + 8]),  ga + 8);
        const bf16_t* gb = Bt + (size_t)(n0 + brow) * K_DIM + k0 + bcol;
        async_load_b128(lds_addr(&Bs[buf][brow][bcol]),      gb);
        async_load_b128(lds_addr(&Bs[buf][brow][bcol + 8]),  gb + 8);
        async_load_b128(lds_addr(&Bs[buf][brow][bcol + 16]), gb + 16);
        async_load_b128(lds_addr(&Bs[buf][brow][bcol + 24]), gb + 24);
    };

    v8f acc[4];
    const v8f vzero = {0.f, 0.f, 0.f, 0.f, 0.f, 0.f, 0.f, 0.f};
#pragma unroll
    for (int t = 0; t < 4; ++t) acc[t] = vzero;

    const int am  = wm * 16 + (lane & 15);
    const int akb = (lane < 16) ? 0 : 8;
    const int bkb = (lane < 16) ? 0 : 8;

    issue_stage(0, 0);
    for (int k = 0; k < NK; ++k) {
        if (k + 1 < NK) {
            issue_stage(k + 1, (k + 1) & 1);
            asm volatile("s_wait_asynccnt 6" ::: "memory");   // stage k landed
        } else {
            asm volatile("s_wait_asynccnt 0" ::: "memory");
        }
        __syncthreads();   // stage k visible to all waves
        const int buf = k & 1;
#pragma unroll
        for (int ks = 0; ks < BK; ks += 32) {
            v8bf a_lo = *(const v8bf*)&As[buf][am][ks + akb];
            v8bf a_hi = *(const v8bf*)&As[buf][am][ks + akb + 16];
            v16bf afrag;
#pragma unroll
            for (int i = 0; i < 8; ++i) { afrag[i] = a_lo[i]; afrag[8 + i] = a_hi[i]; }
#pragma unroll
            for (int t = 0; t < 4; ++t) {
                const int bn = wn * 64 + t * 16 + (lane & 15);
                v8bf b_lo = *(const v8bf*)&Bs[buf][bn][ks + bkb];
                v8bf b_hi = *(const v8bf*)&Bs[buf][bn][ks + bkb + 16];
                v16bf bfrag;
#pragma unroll
                for (int i = 0; i < 8; ++i) { bfrag[i] = b_lo[i]; bfrag[8 + i] = b_hi[i]; }
                acc[t] = __builtin_amdgcn_wmma_f32_16x16x32_bf16(
                    false, afrag, false, bfrag, (short)0, acc[t], false, false);
            }
        }
        __syncthreads();   // done reading buf before stage k+2 overwrites it
    }

    // ---- epilogue ----
    const int mb = m0 + wm * 16 + ((lane < 16) ? 0 : 8);
    const int nn = n0 + wn * 64 + (lane & 15);
#pragma unroll
    for (int t = 0; t < 4; ++t) {
        const int gn = nn + t * 16;
        const float bi = bias[gn];
        const float sc = (MODE == 2) ? scale[gn] : 0.0f;
#pragma unroll
        for (int r = 0; r < 8; ++r) {
            const int gm = mb + r;
            const float v = acc[t][r] + bi;
            const size_t oidx = (size_t)gm * N_DIM + gn;
            if constexpr (MODE == 1) {
                ((float*)Out)[oidx] = v;
            } else if constexpr (MODE == 2) {
                ((float*)Out)[oidx] = resid[oidx] + v * sc;
            } else {
                ((bf16_t*)Out)[oidx] = (bf16_t)gelu_exact(v);
            }
        }
    }
}

// ---------------------------------------------------------------------------
// Host launcher
// ---------------------------------------------------------------------------
extern "C" void kernel_launch(void* const* d_in, const int* in_sizes, int n_in,
                              void* d_out, int out_size, void* d_ws, size_t ws_size,
                              hipStream_t stream) {
    (void)in_sizes; (void)n_in; (void)out_size; (void)ws_size;

    const float* x      = (const float*)d_in[0];
    const float* ln1_g  = (const float*)d_in[1];
    const float* ln1_b  = (const float*)d_in[2];
    const float* w3     = (const float*)d_in[3];
    const float* b3     = (const float*)d_in[4];
    const float* w5     = (const float*)d_in[5];
    const float* b5     = (const float*)d_in[6];
    const float* w7     = (const float*)d_in[7];
    const float* b7     = (const float*)d_in[8];
    const float* wmix   = (const float*)d_in[9];
    const float* bmix   = (const float*)d_in[10];
    const float* cg_w1  = (const float*)d_in[11];
    const float* cg_b1  = (const float*)d_in[12];
    const float* cg_w2  = (const float*)d_in[13];
    const float* cg_b2  = (const float*)d_in[14];
    const float* wout   = (const float*)d_in[15];
    const float* bout   = (const float*)d_in[16];
    const float* ls1    = (const float*)d_in[17];
    const float* ln2_g  = (const float*)d_in[18];
    const float* ln2_b  = (const float*)d_in[19];
    const float* ffn_w1 = (const float*)d_in[20];
    const float* ffn_b1 = (const float*)d_in[21];
    const float* ffn_w2 = (const float*)d_in[22];
    const float* ffn_b2 = (const float*)d_in[23];
    const float* ls2    = (const float*)d_in[24];

    // ---- workspace carve-up ----
    char* ws = (char*)d_ws;
    size_t off = 0;
    auto carve = [&](size_t bytes) -> void* {
        void* p = ws + off;
        off += (bytes + 255) & ~(size_t)255;
        return p;
    };
    bf16_t* h_ln1   = (bf16_t*)carve((size_t)NROWS * DIMC * 2);  // LN1 output
    bf16_t* g_conv  = (bf16_t*)carve((size_t)NROWS * DIMC * 2);  // gelu(conv mix)
    float*  mixed   = (float*) carve((size_t)NROWS * DIMC * 4);  // after wmix GEMM
    bf16_t* a_gate  = (bf16_t*)carve((size_t)NROWS * DIMC * 2);  // mixed*gate, bf16
    float*  y1      = (float*) carve((size_t)NROWS * DIMC * 4);  // x + attn branch
    bf16_t* h_ln2   = (bf16_t*)carve((size_t)NROWS * DIMC * 2);  // LN2 output
    bf16_t* u_ffn   = (bf16_t*)carve((size_t)NROWS * HIDF * 2);  // gelu(ffn1)
    bf16_t* wmix_b  = (bf16_t*)carve((size_t)DIMC * DIMC * 2);   // [N][K] bf16
    bf16_t* wout_t  = (bf16_t*)carve((size_t)DIMC * DIMC * 2);
    bf16_t* fw1_t   = (bf16_t*)carve((size_t)DIMC * HIDF * 2);
    bf16_t* fw2_t   = (bf16_t*)carve((size_t)HIDF * DIMC * 2);
    float*  ct      = (float*) carve((size_t)BATCH * DIMC * 4);
    float*  gate    = (float*) carve((size_t)BATCH * DIMC * 4);

    // ---- 0. weight prep (bf16, [N][K]) ----
    cast_copy_kernel<<<(DIMC * DIMC + 255) / 256, 256, 0, stream>>>(wmix, wmix_b, DIMC * DIMC);
    transpose_cast_kernel<<<dim3(DIMC / 32, DIMC / 32), dim3(32, 8), 0, stream>>>(wout, wout_t, DIMC, DIMC);
    transpose_cast_kernel<<<dim3(HIDF / 32, DIMC / 32), dim3(32, 8), 0, stream>>>(ffn_w1, fw1_t, DIMC, HIDF);
    transpose_cast_kernel<<<dim3(DIMC / 32, HIDF / 32), dim3(32, 8), 0, stream>>>(ffn_w2, fw2_t, HIDF, DIMC);

    // ---- 1. LN1 ----
    ln_rows_kernel<<<NROWS, 256, 0, stream>>>(x, ln1_g, ln1_b, h_ln1);

    // ---- 2. fused depthwise convs + gelu ----
    conv_gelu_kernel<<<(unsigned)(((size_t)NROWS * DIMC) / 256), 256, 0, stream>>>(
        h_ln1, w3, b3, w5, b5, w7, b7, g_conv);

    // ---- 3. mix GEMM: mixed = g_conv @ wmix^T + bmix   [32768 x 512 x 512] ----
    gemm_wmma_kernel<1, DIMC, DIMC><<<dim3(DIMC / 128, NROWS / 64), 256, 0, stream>>>(
        g_conv, wmix_b, bmix, nullptr, nullptr, mixed);

    // ---- 4. channel token + gate MLP + apply gate ----
    ct_mean_kernel<<<dim3(BATCH, DIMC / 256), 256, 0, stream>>>(mixed, ct);
    gate_kernel<<<BATCH, GHID, 0, stream>>>(ct, cg_w1, cg_b1, cg_w2, cg_b2, gate);
    gate_apply_kernel<<<(unsigned)(((size_t)NROWS * DIMC) / 256), 256, 0, stream>>>(
        mixed, gate, a_gate);

    // ---- 5. out GEMM: y1 = x + (a_gate @ wout + bout)*ls1 ----
    gemm_wmma_kernel<2, DIMC, DIMC><<<dim3(DIMC / 128, NROWS / 64), 256, 0, stream>>>(
        a_gate, wout_t, bout, x, ls1, y1);

    // ---- 6. LN2 ----
    ln_rows_kernel<<<NROWS, 256, 0, stream>>>(y1, ln2_g, ln2_b, h_ln2);

    // ---- 7. FFN1: u = gelu(h_ln2 @ ffn_w1 + b1)   [32768 x 2048 x 512] ----
    gemm_wmma_kernel<3, HIDF, DIMC><<<dim3(HIDF / 128, NROWS / 64), 256, 0, stream>>>(
        h_ln2, fw1_t, ffn_b1, nullptr, nullptr, u_ffn);

    // ---- 8. FFN2: out = y1 + (u @ ffn_w2 + b2)*ls2   [32768 x 512 x 2048] ----
    gemm_wmma_kernel<2, DIMC, HIDF><<<dim3(DIMC / 128, NROWS / 64), 256, 0, stream>>>(
        u_ffn, fw2_t, ffn_b2, y1, ls2, (float*)d_out);
}
